// WaveNet_27066883900211
// MI455X (gfx1250) — matrix operand order repaired and checked by reading, not compiled, not run on hardware
//
#include <hip/hip_runtime.h>
#include <hip/hip_bf16.h>

// ---------------------------------------------------------------------------
// WaveNet-vocoder forward for MI455X (gfx1250, wave32, WMMA).
// Main GEMMs run on v_wmma_f32_16x16x32_f16 (f32 accum), 2 M-tiles per wave
// so every weight B-fragment feeds two WMMAs and hazard slots are hidden.
// ---------------------------------------------------------------------------

typedef _Float16 h8   __attribute__((ext_vector_type(8)));
typedef _Float16 v16h __attribute__((ext_vector_type(16)));
typedef float    v8f  __attribute__((ext_vector_type(8)));

#define B_SZ     2
#define T_LEN    32768
#define C_CH     64
#define NCH2     128
#define KMAIN    288          // 192 conv taps + 80 mel + 16 pad
#define KGS      64
#define MELP     96           // mel padded 80 -> 96
#define N_BLOCKS 30
#define SCALE_F  0.316227766016838f   // 10^-0.5

__device__ __forceinline__ float sigmoidf_(float x) { return 1.0f / (1.0f + __expf(-x)); }
__device__ __forceinline__ float swishf_(float x)   { return x * sigmoidf_(x); }

__device__ __forceinline__ h8 h8_zero() {
    h8 z;
#pragma unroll
    for (int i = 0; i < 8; ++i) z[i] = (_Float16)0.0f;
    return z;
}
__device__ __forceinline__ v16h cat16(h8 a, h8 b) {
    v16h r;
#pragma unroll
    for (int i = 0; i < 8; ++i) { r[i] = a[i]; r[i + 8] = b[i]; }
    return r;
}

// ---------------------------------------------------------------------------
// 1) Sinusoidal embedding + 2-layer swish MLP.  One workgroup, 512 threads.
// ---------------------------------------------------------------------------
__global__ __launch_bounds__(512)
void embed_mlp_kernel(const float* __restrict__ snr,
                      const float* __restrict__ eW0, const float* __restrict__ eb0,
                      const float* __restrict__ eW1, const float* __restrict__ eb1,
                      float* __restrict__ embed_out /* [B][512] */) {
    __shared__ float pe[B_SZ * 128];
    __shared__ float e1[B_SZ * 512];
    int tid = threadIdx.x;
    if (tid < B_SZ * 64) {
        int b = tid >> 6, i = tid & 63;
        float denom = __expf(-logf(10000.0f) * (2.0f * (float)i) / 128.0f);
        float c = snr[b] * denom * 4.0f;
        pe[b * 128 + 2 * i]     = __sinf(c);
        pe[b * 128 + 2 * i + 1] = __cosf(c);
    }
    __syncthreads();
    for (int b = 0; b < B_SZ; ++b) {
        float a = eb0[tid];
        for (int k = 0; k < 128; ++k) a += pe[b * 128 + k] * eW0[k * 512 + tid];
        e1[b * 512 + tid] = swishf_(a);
    }
    __syncthreads();
    for (int b = 0; b < B_SZ; ++b) {
        float a = eb1[tid];
        for (int k = 0; k < 512; ++k) a += e1[b * 512 + k] * eW1[k * 512 + tid];
        embed_out[b * 512 + tid] = swishf_(a);
    }
}

// ---------------------------------------------------------------------------
// 2) Per-block fused bias: cbias[b][n] = conv_b[n] + mel_b[n]
//                          + sum_{k,c} (embed@embed_W + embed_b)[b][c] * conv_W[k][c][n]
// ---------------------------------------------------------------------------
__global__ __launch_bounds__(256)
void block_bias_kernel(const float* __restrict__ embed,   // [B][512]
                       const float* __restrict__ embW,    // [512][64]
                       const float* __restrict__ embB,    // [64]
                       const float* __restrict__ convW,   // [3][64][128]
                       const float* __restrict__ convB,   // [128]
                       const float* __restrict__ melB,    // [128]
                       float* __restrict__ cbias)         // [B][128]
{
    __shared__ float eb[B_SZ][C_CH];
    int tid = threadIdx.x;
    if (tid < B_SZ * C_CH) {
        int b = tid >> 6, c = tid & 63;
        float a = embB[c];
        for (int j = 0; j < 512; ++j) a += embed[b * 512 + j] * embW[j * 64 + c];
        eb[b][c] = a;
    }
    __syncthreads();
    int b = tid >> 7, n = tid & 127;
    float a = convB[n] + melB[n];
    for (int k = 0; k < 3; ++k)
#pragma unroll 4
        for (int c = 0; c < 64; ++c)
            a += eb[b][c] * convW[(k * 64 + c) * 128 + n];
    cbias[b * 128 + n] = a;
}

// ---------------------------------------------------------------------------
// 3) Pack per-block weights into f16 B-matrices (transposed: [n][k] so each
//    lane's WMMA B-fragment is two contiguous 16-byte loads).
// ---------------------------------------------------------------------------
__global__ __launch_bounds__(256)
void pack_weights_kernel(const float* __restrict__ convW, const float* __restrict__ melW,
                         const float* __restrict__ resW,  const float* __restrict__ skipW,
                         _Float16* __restrict__ btm, _Float16* __restrict__ btg) {
    int i = blockIdx.x * 256 + threadIdx.x;
    if (i < NCH2 * KMAIN) {
        int n = i / KMAIN, k = i % KMAIN;
        float v = 0.0f;
        if (k < 192)      { int tap = k >> 6, c = k & 63; v = convW[(tap * 64 + c) * 128 + n]; }
        else if (k < 272) { v = melW[(k - 192) * 128 + n]; }
        btm[n * KMAIN + k] = (_Float16)v;
    } else {
        int j = i - NCH2 * KMAIN;
        if (j < NCH2 * KGS) {
            int n = j / KGS, c = j % KGS;
            float v = (n < 64) ? resW[c * 64 + n] : skipW[c * 64 + (n - 64)];
            btg[n * KGS + c] = (_Float16)v;
        }
    }
}

// ---------------------------------------------------------------------------
// 4) x init:  x = swish(signal * proj_W + proj_b), f32 + f16 copies; ctx = 0.
// ---------------------------------------------------------------------------
__global__ __launch_bounds__(256)
void x_init_kernel(const float* __restrict__ signal, const float* __restrict__ projW,
                   const float* __restrict__ projB,
                   float* __restrict__ xf, _Float16* __restrict__ xh,
                   float* __restrict__ ctx) {
    size_t i = (size_t)blockIdx.x * 256 + threadIdx.x;   // over B*T*64
    size_t bt = i >> 6;
    int c = (int)(i & 63);
    float v = swishf_(signal[bt] * projW[c] + projB[c]);
    xf[i] = v;
    xh[i] = (_Float16)v;
    ctx[i] = 0.0f;
}

// ---------------------------------------------------------------------------
// 5) Mel upsampling: two conv_transpose (kernel 32x3, stride 16x1, SAME)+swish.
// ---------------------------------------------------------------------------
__global__ __launch_bounds__(256)
void upsample1_kernel(const float* __restrict__ mel,  // [B][128][80]
                      const float* __restrict__ W,    // [32][3]
                      const float* __restrict__ bias,
                      float* __restrict__ out)        // [B][2048][80]
{
    int i = blockIdx.x * 256 + threadIdx.x;           // B*2048*80
    if (i >= B_SZ * 2048 * 80) return;
    int f = i % 80, t = (i / 80) % 2048, b = i / (80 * 2048);
    float acc = bias[0];
#pragma unroll
    for (int k = 0; k < 32; ++k) {
        int s = t + 8 - k;
        if (s >= 0 && (s & 15) == 0) {
            int src = s >> 4;
            if (src < 128) {
#pragma unroll
                for (int kw = 0; kw < 3; ++kw) {
                    int fi = f + kw - 1;
                    if (fi >= 0 && fi < 80)
                        acc += mel[(b * 128 + src) * 80 + fi] * W[k * 3 + (2 - kw)];
                }
            }
        }
    }
    out[i] = swishf_(acc);
}

__global__ __launch_bounds__(256)
void upsample2_kernel(const float* __restrict__ in,   // [B][2048][80]
                      const float* __restrict__ W,    // [32][3]
                      const float* __restrict__ bias,
                      _Float16* __restrict__ melh)    // [B][T][96] padded
{
    size_t i = (size_t)blockIdx.x * 256 + threadIdx.x; // B*T*96
    int f = (int)(i % MELP);
    int t = (int)((i / MELP) % T_LEN);
    int b = (int)(i / ((size_t)MELP * T_LEN));
    if (f >= 80) { melh[i] = (_Float16)0.0f; return; }
    float acc = bias[0];
#pragma unroll
    for (int k = 0; k < 32; ++k) {
        int s = t + 8 - k;
        if (s >= 0 && (s & 15) == 0) {
            int src = s >> 4;
            if (src < 2048) {
#pragma unroll
                for (int kw = 0; kw < 3; ++kw) {
                    int fi = f + kw - 1;
                    if (fi >= 0 && fi < 80)
                        acc += in[(b * 2048 + src) * 80 + fi] * W[k * 3 + (2 - kw)];
                }
            }
        }
    }
    melh[i] = (_Float16)swishf_(acc);
}

// ---------------------------------------------------------------------------
// 6) Main residual block (launched 30x, double-buffered x).
//    Each wave: 32 time rows (2 M-tiles) x 128 channels; each weight
//    B-fragment feeds two WMMAs (acc0/acc1) -> hazard slots hidden, weight
//    traffic halved.
//    GEMM1: h[32x128] = [x(t-d)|x(t)|x(t+d)|mel] (K=288) @ btm  -> 9x8x2 WMMA
//    gate in-register, g -> LDS -> A-frags
//    GEMM2: [res|skip][32x128] = g (K=64) @ btg                 -> 2x8x2 WMMA
// ---------------------------------------------------------------------------
__global__ __launch_bounds__(128)
void wavenet_block_kernel(const _Float16* __restrict__ xh_in,
                          const float*    __restrict__ xf_in,
                          float*          __restrict__ xf_out,
                          _Float16*       __restrict__ xh_out,
                          float*          __restrict__ ctx,
                          const _Float16* __restrict__ melh,
                          const _Float16* __restrict__ btm,   // [128][288]
                          const _Float16* __restrict__ btg,   // [128][64]
                          const float*    __restrict__ cbias, // [B][128]
                          const float*    __restrict__ res_b,
                          const float*    __restrict__ skip_b,
                          int dil)
{
    __shared__ _Float16 gsh[4][32 * C_CH];   // per-wave 32x64 gate tile (4KB)

    const int wave = threadIdx.x >> 5;
    const int lane = threadIdx.x & 31;
    const int lhi  = lane >> 4;              // 0/1
    const int llo  = lane & 15;

    const int pairIdx = blockIdx.x * 4 + wave;       // tiles of 32 rows
    const int pairsPerBatch = T_LEN / 32;
    const int b  = pairIdx / pairsPerBatch;
    const int t0 = (pairIdx % pairsPerBatch) * 32;

    // gfx1250 prefetch of the dilated tap rows (outside the streaming window)
    {
        int r0 = t0 + lane - dil;
        int r1 = t0 + lane + dil;
        if (r0 >= 0)
            __builtin_prefetch(xh_in + ((size_t)b * T_LEN + r0) * C_CH, 0, 3);
        if (r1 < T_LEN)
            __builtin_prefetch(xh_in + ((size_t)b * T_LEN + r1) * C_CH, 0, 3);
    }

    v8f acc0[8], acc1[8];
#pragma unroll
    for (int n = 0; n < 8; ++n)
#pragma unroll
        for (int r = 0; r < 8; ++r) { acc0[n][r] = 0.0f; acc1[n][r] = 0.0f; }

    // ---- GEMM 1: K = 288 in 9 steps of 32, two M-tiles ----
#pragma unroll
    for (int ks = 0; ks < 9; ++ks) {
        const int kbase = ks * 32;
        v16h af0, af1;
        if (kbase < 192) {                       // dilated conv taps over x
            const int tap = kbase >> 6;
            const int c0  = (kbase & 63) + lhi * 8;
            const int row0 = t0 + llo + (tap - 1) * dil;
            const int row1 = row0 + 16;
            const bool ok0 = (row0 >= 0) && (row0 < T_LEN);
            const bool ok1 = (row1 >= 0) && (row1 < T_LEN);
            const _Float16* b0 = xh_in + (((size_t)b * T_LEN + row0) * C_CH);
            const _Float16* b1 = xh_in + (((size_t)b * T_LEN + row1) * C_CH);
            af0 = cat16(ok0 ? *(const h8*)(b0 + c0)      : h8_zero(),
                        ok0 ? *(const h8*)(b0 + c0 + 16) : h8_zero());
            af1 = cat16(ok1 ? *(const h8*)(b1 + c0)      : h8_zero(),
                        ok1 ? *(const h8*)(b1 + c0 + 16) : h8_zero());
        } else {                                 // mel conditioning (padded 96)
            const int c0  = (kbase - 192) + lhi * 8;
            const _Float16* b0 = melh + (((size_t)b * T_LEN + t0 + llo) * MELP);
            const _Float16* b1 = b0 + 16 * MELP;
            af0 = cat16(*(const h8*)(b0 + c0), *(const h8*)(b0 + c0 + 16));
            af1 = cat16(*(const h8*)(b1 + c0), *(const h8*)(b1 + c0 + 16));
        }
#pragma unroll
        for (int n = 0; n < 8; ++n) {
            const _Float16* bb = btm + ((size_t)(n * 16 + llo) * KMAIN + kbase + lhi * 8);
            v16h bfrag = cat16(((const h8*)bb)[0], ((const h8*)bb)[2]);
            acc0[n] = __builtin_amdgcn_wmma_f32_16x16x32_f16(
                false, af0, false, bfrag, (short)0, acc0[n], false, false);
            acc1[n] = __builtin_amdgcn_wmma_f32_16x16x32_f16(
                false, af1, false, bfrag, (short)0, acc1[n], false, false);
        }
    }

    // ---- bias + gate (tiles j and j+4 share (m,n) lane mapping) ----
    const float* cb = cbias + b * NCH2;
#pragma unroll
    for (int j = 0; j < 4; ++j) {
        const float ct = cb[j * 16 + llo];
        const float cs = cb[64 + j * 16 + llo];
#pragma unroll
        for (int r = 0; r < 8; ++r) {
            const int m = r + lhi * 8;
            float g0 = tanhf(acc0[j][r] + ct) * sigmoidf_(acc0[j + 4][r] + cs);
            float g1 = tanhf(acc1[j][r] + ct) * sigmoidf_(acc1[j + 4][r] + cs);
            gsh[wave][m * C_CH + j * 16 + llo]        = (_Float16)g0;
            gsh[wave][(16 + m) * C_CH + j * 16 + llo] = (_Float16)g1;
        }
    }
    __syncthreads();

    // ---- GEMM 2: [res|skip] = g @ btg, K = 64 in 2 steps, two M-tiles ----
    v8f s0[8], s1[8];
#pragma unroll
    for (int n = 0; n < 8; ++n)
#pragma unroll
        for (int r = 0; r < 8; ++r) { s0[n][r] = 0.0f; s1[n][r] = 0.0f; }

#pragma unroll
    for (int ks = 0; ks < 2; ++ks) {
        const int kbase = ks * 32;
        const _Float16* g0 = &gsh[wave][llo * C_CH + kbase + lhi * 8];
        const _Float16* g1 = &gsh[wave][(16 + llo) * C_CH + kbase + lhi * 8];
        v16h ag0 = cat16(*(const h8*)g0, *(const h8*)(g0 + 16));
        v16h ag1 = cat16(*(const h8*)g1, *(const h8*)(g1 + 16));
#pragma unroll
        for (int n = 0; n < 8; ++n) {
            const _Float16* bb = btg + ((size_t)(n * 16 + llo) * KGS + kbase + lhi * 8);
            v16h bfrag = cat16(((const h8*)bb)[0], ((const h8*)bb)[2]);
            s0[n] = __builtin_amdgcn_wmma_f32_16x16x32_f16(
                false, ag0, false, bfrag, (short)0, s0[n], false, false);
            s1[n] = __builtin_amdgcn_wmma_f32_16x16x32_f16(
                false, ag1, false, bfrag, (short)0, s1[n], false, false);
        }
    }

    // ---- residual (+x) and skip (+=ctx), coalesced per-lane stores ----
#pragma unroll
    for (int j = 0; j < 4; ++j) {
        const int col = j * 16 + llo;
        const float rb = res_b[col];
        const float sb = skip_b[col];
#pragma unroll
        for (int r = 0; r < 8; ++r) {
            const int m = r + lhi * 8;
            const size_t xi0 = ((size_t)b * T_LEN + t0 + m) * C_CH + col;
            const size_t xi1 = ((size_t)b * T_LEN + t0 + 16 + m) * C_CH + col;
            float xn0 = s0[j][r] + rb + xf_in[xi0];
            float xn1 = s1[j][r] + rb + xf_in[xi1];
            xf_out[xi0] = xn0;           xf_out[xi1] = xn1;
            xh_out[xi0] = (_Float16)xn0; xh_out[xi1] = (_Float16)xn1;
            ctx[xi0] += s0[j + 4][r] + sb;
            ctx[xi1] += s1[j + 4][r] + sb;
        }
    }
}

// ---------------------------------------------------------------------------
// 7) Output head: tanh(swish(ctx*SCALE @ ctx_W + ctx_b) @ out_W + out_b)
// ---------------------------------------------------------------------------
__global__ __launch_bounds__(256)
void head_kernel(const float* __restrict__ ctx, const float* __restrict__ ctxW,
                 const float* __restrict__ ctxB, const float* __restrict__ outW,
                 const float* __restrict__ outB, float* __restrict__ out) {
    __shared__ float Ws[64 * 64];
    __shared__ float wb[64];
    __shared__ float wo[64];
    int tid = threadIdx.x;
    for (int i = tid; i < 64 * 64; i += 256) Ws[i] = ctxW[i];
    if (tid < 64) { wb[tid] = ctxB[tid]; wo[tid] = outW[tid]; }
    __syncthreads();

    size_t idx = (size_t)blockIdx.x * 256 + tid;       // over B*T
    float c[64];
#pragma unroll
    for (int k = 0; k < 64; ++k) c[k] = ctx[idx * 64 + k] * SCALE_F;
    float o = outB[0];
    for (int j = 0; j < 64; ++j) {
        float a = wb[j];
#pragma unroll
        for (int k = 0; k < 64; ++k) a += c[k] * Ws[k * 64 + j];
        o += swishf_(a) * wo[j];
    }
    out[idx] = tanhf(o);
}

// ---------------------------------------------------------------------------
// Launcher
// ---------------------------------------------------------------------------
extern "C" void kernel_launch(void* const* d_in, const int* in_sizes, int n_in,
                              void* d_out, int out_size, void* d_ws, size_t ws_size,
                              hipStream_t stream) {
    (void)in_sizes; (void)n_in; (void)out_size; (void)ws_size;

    // input layout: setup_inputs() insertion order, params flattened in order
    const float* signal = (const float*)d_in[0];
    const float* snr    = (const float*)d_in[1];
    const float* mel    = (const float*)d_in[2];
    const float* projW  = (const float*)d_in[3];
    const float* projB  = (const float*)d_in[4];
    const float* eW0    = (const float*)d_in[5];
    const float* eW1    = (const float*)d_in[6];
    const float* eb0    = (const float*)d_in[7];
    const float* eb1    = (const float*)d_in[8];
    const float* upW0   = (const float*)d_in[9];
    const float* upW1   = (const float*)d_in[10];
    const float* upb0   = (const float*)d_in[11];
    const float* upb1   = (const float*)d_in[12];
    const int BLK0 = 13;   // blk param i: BLK0 + blk*10 + i
    const float* ctxW = (const float*)d_in[BLK0 + N_BLOCKS * 10 + 0];
    const float* ctxB = (const float*)d_in[BLK0 + N_BLOCKS * 10 + 1];
    const float* outW = (const float*)d_in[BLK0 + N_BLOCKS * 10 + 2];
    const float* outB = (const float*)d_in[BLK0 + N_BLOCKS * 10 + 3];

    // workspace carve-up (256-byte aligned)
    char* ws = (char*)d_ws;
    size_t off = 0;
    auto carve = [&](size_t bytes) -> char* {
        char* p = ws + off;
        off = (off + bytes + 255) & ~(size_t)255;
        return p;
    };
    float*    xf[2];  xf[0] = (float*)carve((size_t)B_SZ * T_LEN * C_CH * 4);
                      xf[1] = (float*)carve((size_t)B_SZ * T_LEN * C_CH * 4);
    _Float16* xh[2];  xh[0] = (_Float16*)carve((size_t)B_SZ * T_LEN * C_CH * 2);
                      xh[1] = (_Float16*)carve((size_t)B_SZ * T_LEN * C_CH * 2);
    float*    ctx    = (float*)carve((size_t)B_SZ * T_LEN * C_CH * 4);
    _Float16* melh   = (_Float16*)carve((size_t)B_SZ * T_LEN * MELP * 2);
    float*    m1     = (float*)carve((size_t)B_SZ * 2048 * 80 * 4);
    float*    embed  = (float*)carve((size_t)B_SZ * 512 * 4);
    float*    cbias  = (float*)carve((size_t)N_BLOCKS * B_SZ * NCH2 * 4);
    _Float16* btm    = (_Float16*)carve((size_t)N_BLOCKS * NCH2 * KMAIN * 2);
    _Float16* btg    = (_Float16*)carve((size_t)N_BLOCKS * NCH2 * KGS * 2);

    // embedding MLP
    embed_mlp_kernel<<<1, 512, 0, stream>>>(snr, eW0, eb0, eW1, eb1, embed);

    // per-block fused bias + weight packing
    for (int blk = 0; blk < N_BLOCKS; ++blk) {
        const float* embW  = (const float*)d_in[BLK0 + blk * 10 + 0];
        const float* embB  = (const float*)d_in[BLK0 + blk * 10 + 1];
        const float* melW  = (const float*)d_in[BLK0 + blk * 10 + 2];
        const float* melB  = (const float*)d_in[BLK0 + blk * 10 + 3];
        const float* convW = (const float*)d_in[BLK0 + blk * 10 + 4];
        const float* convB = (const float*)d_in[BLK0 + blk * 10 + 5];
        const float* resW  = (const float*)d_in[BLK0 + blk * 10 + 6];
        const float* skipW = (const float*)d_in[BLK0 + blk * 10 + 8];
        block_bias_kernel<<<1, 256, 0, stream>>>(embed, embW, embB, convW, convB, melB,
                                                 cbias + (size_t)blk * B_SZ * NCH2);
        int nelem = NCH2 * KMAIN + NCH2 * KGS;   // 45056
        pack_weights_kernel<<<(nelem + 255) / 256, 256, 0, stream>>>(
            convW, melW, resW, skipW,
            btm + (size_t)blk * NCH2 * KMAIN, btg + (size_t)blk * NCH2 * KGS);
    }

    // x init (+ ctx zero)
    {
        size_t n = (size_t)B_SZ * T_LEN * C_CH;
        x_init_kernel<<<(unsigned)(n / 256), 256, 0, stream>>>(signal, projW, projB,
                                                               xf[0], xh[0], ctx);
    }

    // mel upsampling
    upsample1_kernel<<<(B_SZ * 2048 * 80 + 255) / 256, 256, 0, stream>>>(mel, upW0, upb0, m1);
    {
        size_t n = (size_t)B_SZ * T_LEN * MELP;
        upsample2_kernel<<<(unsigned)(n / 256), 256, 0, stream>>>(m1, upW1, upb1, melh);
    }

    // 30 residual blocks (sequential, double-buffered x)
    for (int blk = 0; blk < N_BLOCKS; ++blk) {
        int src = blk & 1, dst = src ^ 1;
        int dil = 1 << (blk % 10);
        const float* res_b  = (const float*)d_in[BLK0 + blk * 10 + 7];
        const float* skip_b = (const float*)d_in[BLK0 + blk * 10 + 9];
        wavenet_block_kernel<<<(B_SZ * T_LEN / 32) / 4, 128, 0, stream>>>(
            xh[src], xf[src], xf[dst], xh[dst], ctx, melh,
            btm + (size_t)blk * NCH2 * KMAIN, btg + (size_t)blk * NCH2 * KGS,
            cbias + (size_t)blk * B_SZ * NCH2, res_b, skip_b, dil);
    }

    // output head
    head_kernel<<<(B_SZ * T_LEN) / 256, 256, 0, stream>>>(ctx, ctxW, ctxB, outW, outB,
                                                          (float*)d_out);
}
